// DualCausalMambaBlock_43250320670995
// MI455X (gfx1250) — compile-verified, matrix operand order checked by reading, not run on hardware
//
#include <hip/hip_runtime.h>
#include <hip/hip_bf16.h>

// ---------------------------------------------------------------------------
// Problem constants (match reference)
// ---------------------------------------------------------------------------
constexpr int B_SZ    = 2;
constexpr int L_SEQ   = 2048;
constexpr int D_MODEL = 1024;
constexpr int D_INNER = 2048;
constexpr int D_STATE = 16;
constexpr int D_CONV  = 4;
constexpr int DT_RANK = 64;
constexpr int M_ROWS  = B_SZ * L_SEQ;            // 4096
constexpr int XP_COLS = DT_RANK + 2 * D_STATE;   // 96
constexpr int CHUNK_L = 64;                      // scan chunk length
constexpr int N_CHUNK = L_SEQ / CHUNK_L;         // 32

typedef __bf16 v16bf __attribute__((ext_vector_type(16)));
typedef float  v8f   __attribute__((ext_vector_type(8)));
typedef unsigned int u32x4 __attribute__((ext_vector_type(4)));
typedef int i32x4 __attribute__((ext_vector_type(4)));
typedef int i32x8 __attribute__((ext_vector_type(8)));

#if defined(__gfx1250__) && __has_builtin(__builtin_amdgcn_tensor_load_to_lds) && \
    __has_builtin(__builtin_amdgcn_s_wait_tensorcnt)
#define USE_TDM 1
#endif

static __device__ __forceinline__ __bf16 f2bf(float f) {
    unsigned u = __builtin_bit_cast(unsigned, f);
    unsigned r = (u + 0x7FFFu + ((u >> 16) & 1u)) >> 16;
    unsigned short s = (unsigned short)r;
    return __builtin_bit_cast(__bf16, s);
}
static __device__ __forceinline__ float bf2f(__bf16 b) {
    unsigned u = (unsigned)__builtin_bit_cast(unsigned short, b) << 16;
    return __builtin_bit_cast(float, u);
}
static __device__ __forceinline__ float sigmoidf_fast(float x) {
    return 1.0f / (1.0f + __expf(-x));
}
static __device__ __forceinline__ float softplusf(float x) {
    return (x > 20.0f) ? x : __logf(1.0f + __expf(x));
}

#ifdef USE_TDM
// 32-bit LDS byte offset of a __shared__ object
static __device__ __forceinline__ unsigned lds_off(const void* p) {
    return (unsigned)(unsigned long long)
        (__attribute__((address_space(3))) const char*)p;
}
struct TdmDesc { u32x4 g0; i32x8 g1; };
// 2D tensor (rows x d0 elems, 2-byte elems), tile (tile_d1 rows x tile_d0 elems)
static __device__ __forceinline__ TdmDesc tdm_make_2d(
        const void* gaddr, unsigned lds_byte, int tensor_d0, int tensor_d1,
        int tile_d0, int tile_d1, long long row_stride_elems) {
    unsigned long long ga = (unsigned long long)gaddr;
    TdmDesc t;
    t.g0.x = 0x1u;                                             // count=1
    t.g0.y = lds_byte;                                         // lds_addr
    t.g0.z = (unsigned)(ga & 0xFFFFFFFFull);                   // global_addr lo
    t.g0.w = (unsigned)((ga >> 32) & 0x1FFFFFFull) | (2u << 30); // hi | type=2
    t.g1[0] = (1 << 16);                                       // data_size=2B
    t.g1[1] = (tensor_d0 & 0xFFFF) << 16;                      // dim0 lo16
    t.g1[2] = ((tensor_d0 >> 16) & 0xFFFF) | ((tensor_d1 & 0xFFFF) << 16);
    t.g1[3] = ((tensor_d1 >> 16) & 0xFFFF) | ((tile_d0 & 0xFFFF) << 16);
    t.g1[4] = (tile_d1 & 0xFFFF);                              // tile_dim1
    t.g1[5] = (int)(row_stride_elems & 0xFFFFFFFFll);          // dim0 stride lo
    t.g1[6] = (int)((row_stride_elems >> 32) & 0xFFFFll);      // stride hi16
    t.g1[7] = 0;
    return t;
}
static __device__ __forceinline__ void tdm_issue(const TdmDesc& t) {
    i32x4 z4 = {0, 0, 0, 0};
#if __clang_major__ >= 23
    i32x8 z8 = {0, 0, 0, 0, 0, 0, 0, 0};
    __builtin_amdgcn_tensor_load_to_lds(t.g0, t.g1, z4, z4, z8, 0);
#else
    __builtin_amdgcn_tensor_load_to_lds(t.g0, t.g1, z4, z4, 0);
#endif
}
#endif // USE_TDM

// ---------------------------------------------------------------------------
// fp32 -> bf16 converters
// ---------------------------------------------------------------------------
__global__ void __launch_bounds__(256)
k_f32_to_bf16(const float* __restrict__ src, __bf16* __restrict__ dst, long n) {
    long i = (long)blockIdx.x * blockDim.x + threadIdx.x;
    long stride = (long)gridDim.x * blockDim.x;
    for (; i < n; i += stride) dst[i] = f2bf(src[i]);
}
__global__ void __launch_bounds__(256)
k_slice_to_bf16(const float* __restrict__ src, __bf16* __restrict__ dst,
                int rows, int cols, int ld) {
    long n = (long)rows * cols;
    long i = (long)blockIdx.x * blockDim.x + threadIdx.x;
    long stride = (long)gridDim.x * blockDim.x;
    for (; i < n; i += stride) {
        int r = (int)(i / cols), c = (int)(i % cols);
        dst[i] = f2bf(src[(long)r * ld + c]);
    }
}

// ---------------------------------------------------------------------------
// bf16 WMMA GEMM:  C[M x N] = A[M x K] * W^T  (W stored N x K)
// 128x128 workgroup tile, 8 waves (2x4), each wave 64x32 -> 8 WMMA / K-step.
// K tiled by 32; tiles DMA'd into double-buffered LDS by the TDM.
// act: 0 = none, 1 = softplus(acc + bias[n])
// ---------------------------------------------------------------------------
#define BM 128
#define BN 128
#define BK 32

__global__ void __launch_bounds__(256)
k_gemm_bf16_wmma(const __bf16* __restrict__ A,   // M x K
                 const __bf16* __restrict__ W,   // N x K
                 float* __restrict__ C,          // M x ldc
                 int M, int N, int K, int ldc,
                 const float* __restrict__ bias, int act) {
    __shared__ __align__(32) __bf16 Asm[2][BM * BK];   // 2 x 8 KB
    __shared__ __align__(32) __bf16 Bsm[2][BN * BK];   // 2 x 8 KB

    const int tid  = threadIdx.x;
    const int lane = tid & 31;
    const int wave = tid >> 5;            // 0..7
    const int wm   = wave >> 2;           // 0..1 -> 64-row slab
    const int wn   = wave & 3;            // 0..3 -> 32-col slab
    const int blockM = blockIdx.y * BM;
    const int blockN = blockIdx.x * BN;
    const int laneHalf = lane >> 4;
    const int laneMod  = lane & 15;

    v8f acc[4][2];
    #pragma unroll
    for (int i = 0; i < 4; ++i)
        #pragma unroll
        for (int j = 0; j < 2; ++j)
            acc[i][j] = (v8f){0.f,0.f,0.f,0.f,0.f,0.f,0.f,0.f};

#ifdef USE_TDM
    // --- TDM double-buffered pipeline: DMA tile k+1 while WMMA-ing tile k ---
    TdmDesc dA = tdm_make_2d(A + (long)blockM * K, lds_off(&Asm[0][0]),
                             K, M - blockM, BK, BM, K);
    TdmDesc dB = tdm_make_2d(W + (long)blockN * K, lds_off(&Bsm[0][0]),
                             K, N - blockN, BK, BN, K);
    const unsigned aXor = lds_off(&Asm[1][0]) ^ lds_off(&Asm[0][0]);
    const unsigned bXor = lds_off(&Bsm[1][0]) ^ lds_off(&Bsm[0][0]);
    if (wave == 0) { tdm_issue(dA); tdm_issue(dB); }
    int p = 0;
    for (int kk = 0; kk < K; kk += BK) {
        if (wave == 0) __builtin_amdgcn_s_wait_tensorcnt(0);
        __syncthreads();                     // tile p ready for everyone
        if (wave == 0 && kk + BK < K) {      // DMA next tile into buffer 1-p
            dA.g0.z += BK * 2; dA.g0.y ^= aXor;
            dB.g0.z += BK * 2; dB.g0.y ^= bXor;
            tdm_issue(dA); tdm_issue(dB);
        }
        v16bf aFrag[4], bFrag[2];
        #pragma unroll
        for (int mi = 0; mi < 4; ++mi) {
            int row = wm * 64 + mi * 16 + laneMod;
            aFrag[mi] = *(const v16bf*)&Asm[p][row * BK + laneHalf * 16];
        }
        #pragma unroll
        for (int ni = 0; ni < 2; ++ni) {
            int row = wn * 32 + ni * 16 + laneMod;
            bFrag[ni] = *(const v16bf*)&Bsm[p][row * BK + laneHalf * 16];
        }
        #pragma unroll
        for (int mi = 0; mi < 4; ++mi)
            #pragma unroll
            for (int ni = 0; ni < 2; ++ni)
                acc[mi][ni] = __builtin_amdgcn_wmma_f32_16x16x32_bf16(
                    false, aFrag[mi], false, bFrag[ni],
                    (short)0, acc[mi][ni], false, false);
        p ^= 1;
    }
#else
    // --- fallback: cooperative register staging into single LDS buffer ---
    const int sr = tid >> 1;               // 128 rows, 2 threads/row
    const int sc = (tid & 1) * 16;         // 16 halves (32 B) each
    const int gmA = blockM + sr;
    const int gnB = (blockN + sr < N) ? (blockN + sr) : (N - 1);
    for (int kk = 0; kk < K; kk += BK) {
        const uint4* sa = (const uint4*)(A + (long)gmA * K + kk + sc);
        const uint4* sb = (const uint4*)(W + (long)gnB * K + kk + sc);
        uint4 a0 = sa[0], a1 = sa[1], b0 = sb[0], b1 = sb[1];
        if (kk + BK < K) __builtin_prefetch(A + (long)gmA * K + kk + BK, 0, 0);
        *(uint4*)&Asm[0][sr * BK + sc]     = a0;
        *(uint4*)&Asm[0][sr * BK + sc + 8] = a1;
        *(uint4*)&Bsm[0][sr * BK + sc]     = b0;
        *(uint4*)&Bsm[0][sr * BK + sc + 8] = b1;
        __syncthreads();
        v16bf aFrag[4], bFrag[2];
        #pragma unroll
        for (int mi = 0; mi < 4; ++mi) {
            int row = wm * 64 + mi * 16 + laneMod;
            aFrag[mi] = *(const v16bf*)&Asm[0][row * BK + laneHalf * 16];
        }
        #pragma unroll
        for (int ni = 0; ni < 2; ++ni) {
            int row = wn * 32 + ni * 16 + laneMod;
            bFrag[ni] = *(const v16bf*)&Bsm[0][row * BK + laneHalf * 16];
        }
        #pragma unroll
        for (int mi = 0; mi < 4; ++mi)
            #pragma unroll
            for (int ni = 0; ni < 2; ++ni)
                acc[mi][ni] = __builtin_amdgcn_wmma_f32_16x16x32_bf16(
                    false, aFrag[mi], false, bFrag[ni],
                    (short)0, acc[mi][ni], false, false);
        __syncthreads();
    }
#endif

    // ---- epilogue / store ----
    #pragma unroll
    for (int mi = 0; mi < 4; ++mi) {
        #pragma unroll
        for (int ni = 0; ni < 2; ++ni) {
            int gn = blockN + wn * 32 + ni * 16 + laneMod;
            if (gn >= N) continue;
            #pragma unroll
            for (int r = 0; r < 8; ++r) {
                int gm = blockM + wm * 64 + mi * 16 + r + 8 * laneHalf;
                float v = acc[mi][ni][r];
                if (act == 1) v = softplusf(v + bias[gn]);
                C[(long)gm * ldc + gn] = v;
            }
        }
    }
}

// ---------------------------------------------------------------------------
// Depthwise causal conv (K=4) + bias + SiLU -> bf16 x
// ---------------------------------------------------------------------------
__global__ void __launch_bounds__(256)
k_conv_silu(const float* __restrict__ xz,     // (B, L, 2*D_INNER)
            const float* __restrict__ conv_w, // (D_INNER, 4)
            const float* __restrict__ conv_b, // (D_INNER)
            __bf16* __restrict__ x_bf) {      // (B, L, D_INNER)
    long idx = (long)blockIdx.x * blockDim.x + threadIdx.x;
    long total = (long)B_SZ * L_SEQ * D_INNER;
    if (idx >= total) return;
    int d = (int)(idx % D_INNER);
    int l = (int)((idx / D_INNER) % L_SEQ);
    int b = (int)(idx / ((long)D_INNER * L_SEQ));

    float acc = conv_b[d];
    const float w0 = conv_w[d * D_CONV + 0];
    const float w1 = conv_w[d * D_CONV + 1];
    const float w2 = conv_w[d * D_CONV + 2];
    const float w3 = conv_w[d * D_CONV + 3];
    const long rowBase = ((long)b * L_SEQ) * (2 * D_INNER) + d;
    const long rs = 2 * D_INNER;
    if (l >= 3) acc += xz[rowBase + (long)(l - 3) * rs] * w0;
    if (l >= 2) acc += xz[rowBase + (long)(l - 2) * rs] * w1;
    if (l >= 1) acc += xz[rowBase + (long)(l - 1) * rs] * w2;
    acc += xz[rowBase + (long)l * rs] * w3;
    x_bf[idx] = f2bf(acc * sigmoidf_fast(acc));
}

// ---------------------------------------------------------------------------
// Chunk-parallel selective scan.
// Pass 1: per chunk, zero-init local scan -> P (decay product), S (local state)
// ---------------------------------------------------------------------------
__global__ void __launch_bounds__(256)
k_scan_pass1(const float* __restrict__ dt, const __bf16* __restrict__ x_bf,
             const float* __restrict__ x_p, const float* __restrict__ A_log,
             float* __restrict__ P, float* __restrict__ S) {
    __shared__ float sB[CHUNK_L * D_STATE];           // 4 KB

    const int dblk = blockIdx.x & 7;                  // D_INNER/256 = 8
    const int c    = (blockIdx.x >> 3) & (N_CHUNK - 1);
    const int b    = blockIdx.x >> 3 >> 5;            // / N_CHUNK
    const int d    = dblk * 256 + threadIdx.x;
    const int t0   = c * CHUNK_L;

    for (int i = threadIdx.x; i < CHUNK_L * D_STATE; i += 256) {
        int t = i >> 4, n = i & 15;
        sB[i] = x_p[((long)b * L_SEQ + t0 + t) * XP_COLS + DT_RANK + n];
    }
    __syncthreads();

    float Acoef[D_STATE], Pn[D_STATE], h[D_STATE];
    #pragma unroll
    for (int n = 0; n < D_STATE; ++n) {
        Acoef[n] = -__expf(A_log[(long)d * D_STATE + n]);
        Pn[n] = 1.0f; h[n] = 0.0f;
    }
    const long chanBase = ((long)b * L_SEQ + t0) * D_INNER + d;
    for (int t = 0; t < CHUNK_L; ++t) {
        float dtv = dt[chanBase + (long)t * D_INNER];
        float xv  = bf2f(x_bf[chanBase + (long)t * D_INNER]);
        float dx  = dtv * xv;
        #pragma unroll
        for (int n = 0; n < D_STATE; ++n) {
            float dAe = __expf(dtv * Acoef[n]);
            Pn[n] *= dAe;
            h[n] = dAe * h[n] + dx * sB[t * D_STATE + n];
        }
    }
    long base = (((long)b * N_CHUNK + c) * D_INNER + d) * D_STATE;
    #pragma unroll
    for (int n = 0; n < D_STATE; ++n) { P[base + n] = Pn[n]; S[base + n] = h[n]; }
}

// Pass 2: 32-step scan over chunk boundaries; P overwritten with H_init[c].
__global__ void __launch_bounds__(256)
k_scan_pass2(float* __restrict__ P, const float* __restrict__ S) {
    long idx = (long)blockIdx.x * blockDim.x + threadIdx.x;   // B*D_INNER*16
    long dn  = idx % ((long)D_INNER * D_STATE);
    long b   = idx / ((long)D_INNER * D_STATE);
    float H = 0.0f;
    for (int c = 0; c < N_CHUNK; ++c) {
        long off = (b * N_CHUNK + c) * (long)D_INNER * D_STATE + dn;
        float p = P[off], s = S[off];
        P[off] = H;                    // state entering chunk c
        H = p * H + s;
    }
}

// Pass 3: re-scan each chunk from H_init, emit gated bf16 output.
__global__ void __launch_bounds__(256)
k_scan_pass3(const float* __restrict__ dt, const __bf16* __restrict__ x_bf,
             const float* __restrict__ xz, const float* __restrict__ x_p,
             const float* __restrict__ A_log, const float* __restrict__ D_param,
             const float* __restrict__ Hinit, __bf16* __restrict__ yg_bf) {
    __shared__ float sBC[CHUNK_L * 2 * D_STATE];      // 8 KB

    const int dblk = blockIdx.x & 7;
    const int c    = (blockIdx.x >> 3) & (N_CHUNK - 1);
    const int b    = blockIdx.x >> 3 >> 5;
    const int d    = dblk * 256 + threadIdx.x;
    const int t0   = c * CHUNK_L;

    for (int i = threadIdx.x; i < CHUNK_L * 2 * D_STATE; i += 256) {
        int t = i >> 5, nn = i & 31;
        sBC[i] = x_p[((long)b * L_SEQ + t0 + t) * XP_COLS + DT_RANK + nn];
    }
    __syncthreads();

    float Acoef[D_STATE], h[D_STATE];
    long hBase = (((long)b * N_CHUNK + c) * D_INNER + d) * D_STATE;
    #pragma unroll
    for (int n = 0; n < D_STATE; ++n) {
        Acoef[n] = -__expf(A_log[(long)d * D_STATE + n]);
        h[n] = Hinit[hBase + n];
    }
    const float Dd = D_param[d];
    const long chanBase = ((long)b * L_SEQ + t0) * D_INNER + d;
    const long zBase    = ((long)b * L_SEQ + t0) * (2 * D_INNER) + D_INNER + d;

    for (int t = 0; t < CHUNK_L; ++t) {
        float dtv = dt[chanBase + (long)t * D_INNER];
        float xv  = bf2f(x_bf[chanBase + (long)t * D_INNER]);
        float zv  = xz[zBase + (long)t * (2 * D_INNER)];
        float dx  = dtv * xv;
        float y = 0.0f;
        #pragma unroll
        for (int n = 0; n < D_STATE; ++n) {
            float dAe = __expf(dtv * Acoef[n]);
            h[n] = dAe * h[n] + dx * sBC[t * 2 * D_STATE + n];
            y += h[n] * sBC[t * 2 * D_STATE + D_STATE + n];
        }
        y += Dd * xv;
        y *= zv * sigmoidf_fast(zv);
        yg_bf[chanBase + (long)t * D_INNER] = f2bf(y);
    }
}

// ---------------------------------------------------------------------------
// Host-side orchestration
// ---------------------------------------------------------------------------
extern "C" void kernel_launch(void* const* d_in, const int* in_sizes, int n_in,
                              void* d_out, int out_size, void* d_ws, size_t ws_size,
                              hipStream_t stream) {
    const float* u       = (const float*)d_in[0];
    const float* W_in    = (const float*)d_in[1];
    const float* W_out   = (const float*)d_in[2];
    const float* conv_w  = (const float*)d_in[3];
    const float* conv_b  = (const float*)d_in[4];
    const float* W_x     = (const float*)d_in[5];
    const float* W_dt    = (const float*)d_in[6];
    const float* b_dt    = (const float*)d_in[7];
    const float* A_log   = (const float*)d_in[8];
    const float* D_param = (const float*)d_in[9];
    float* out = (float*)d_out;

    size_t off = 0;
    char* base = (char*)d_ws;
    auto alloc = [&](size_t bytes) -> void* {
        void* p = base + off;
        off = (off + bytes + 255) & ~(size_t)255;
        return p;
    };
    __bf16* u_bf    = (__bf16*)alloc((size_t)M_ROWS * D_MODEL * 2);
    __bf16* Win_bf  = (__bf16*)alloc((size_t)2 * D_INNER * D_MODEL * 2);
    __bf16* Wx_bf   = (__bf16*)alloc((size_t)XP_COLS * D_INNER * 2);
    __bf16* Wdt_bf  = (__bf16*)alloc((size_t)D_INNER * DT_RANK * 2);
    __bf16* Wout_bf = (__bf16*)alloc((size_t)D_MODEL * D_INNER * 2);
    float*  xz      = (float*) alloc((size_t)M_ROWS * 2 * D_INNER * 4);
    __bf16* x_bf    = (__bf16*)alloc((size_t)M_ROWS * D_INNER * 2);
    float*  x_p     = (float*) alloc((size_t)M_ROWS * XP_COLS * 4);
    __bf16* dtr_bf  = (__bf16*)alloc((size_t)M_ROWS * DT_RANK * 2);
    float*  dt      = (float*) alloc((size_t)M_ROWS * D_INNER * 4);
    __bf16* yg_bf   = (__bf16*)alloc((size_t)M_ROWS * D_INNER * 2);
    float*  Pbuf    = (float*) alloc((size_t)B_SZ * N_CHUNK * D_INNER * D_STATE * 4);
    float*  Sbuf    = (float*) alloc((size_t)B_SZ * N_CHUNK * D_INNER * D_STATE * 4);

    auto cvt = [&](const float* s, __bf16* d, long n) {
        int grid = (int)((n + 255) / 256); if (grid > 16384) grid = 16384;
        k_f32_to_bf16<<<grid, 256, 0, stream>>>(s, d, n);
    };
    cvt(u,     u_bf,    (long)M_ROWS * D_MODEL);
    cvt(W_in,  Win_bf,  (long)2 * D_INNER * D_MODEL);
    cvt(W_x,   Wx_bf,   (long)XP_COLS * D_INNER);
    cvt(W_dt,  Wdt_bf,  (long)D_INNER * DT_RANK);
    cvt(W_out, Wout_bf, (long)D_MODEL * D_INNER);

    // GEMM1: xz = u @ W_in^T
    k_gemm_bf16_wmma<<<dim3(2 * D_INNER / BN, M_ROWS / BM), 256, 0, stream>>>(
        u_bf, Win_bf, xz, M_ROWS, 2 * D_INNER, D_MODEL, 2 * D_INNER, nullptr, 0);

    // conv + SiLU
    {
        long total = (long)M_ROWS * D_INNER;
        k_conv_silu<<<(int)((total + 255) / 256), 256, 0, stream>>>(
            xz, conv_w, conv_b, x_bf);
    }

    // GEMM2: x_p = x @ W_x^T (N=96, OOB rows zero-filled by TDM / guarded stores)
    k_gemm_bf16_wmma<<<dim3((XP_COLS + BN - 1) / BN, M_ROWS / BM), 256, 0, stream>>>(
        x_bf, Wx_bf, x_p, M_ROWS, XP_COLS, D_INNER, XP_COLS, nullptr, 0);

    // dt_r slice -> bf16
    {
        long total = (long)M_ROWS * DT_RANK;
        k_slice_to_bf16<<<(int)((total + 255) / 256), 256, 0, stream>>>(
            x_p, dtr_bf, M_ROWS, DT_RANK, XP_COLS);
    }

    // GEMM3: dt = softplus(dt_r @ W_dt^T + b_dt)
    k_gemm_bf16_wmma<<<dim3(D_INNER / BN, M_ROWS / BM), 256, 0, stream>>>(
        dtr_bf, Wdt_bf, dt, M_ROWS, D_INNER, DT_RANK, D_INNER, b_dt, 1);

    // chunk-parallel scan
    {
        int blocks13 = B_SZ * N_CHUNK * (D_INNER / 256);      // 512
        k_scan_pass1<<<blocks13, 256, 0, stream>>>(dt, x_bf, x_p, A_log, Pbuf, Sbuf);
        int total2 = B_SZ * D_INNER * D_STATE;                // 65536
        k_scan_pass2<<<total2 / 256, 256, 0, stream>>>(Pbuf, Sbuf);
        k_scan_pass3<<<blocks13, 256, 0, stream>>>(
            dt, x_bf, xz, x_p, A_log, D_param, Pbuf, yg_bf);
    }

    // GEMM4: out = yg @ W_out^T
    k_gemm_bf16_wmma<<<dim3(D_MODEL / BN, M_ROWS / BM), 256, 0, stream>>>(
        yg_bf, Wout_bf, out, M_ROWS, D_MODEL, D_INNER, D_MODEL, nullptr, 0);
}
